// BlockLocalSelfAttention_66537633350056
// MI455X (gfx1250) — matrix operand order repaired and checked by reading, not compiled, not run on hardware
//
#include <hip/hip_runtime.h>

typedef __attribute__((ext_vector_type(16))) _Float16 v16h;
typedef __attribute__((ext_vector_type(8)))  _Float16 v8h;
typedef __attribute__((ext_vector_type(8)))  float    v8f;

#define TSEQ  4096
#define DH    64
#define BQ    128
#define NEGF  (-1.0e30f)
#define SCALE 0.125f   // 1/sqrt(64)

union HV { v16h v; v8h h[2]; };

// DPP row-rotate max: reduces within each 16-lane row in pure VALU.
template <int CTRL>
__device__ __forceinline__ float rowmax_step(float x) {
  int xi = __float_as_int(x);
  int yi = __builtin_amdgcn_update_dpp(xi, xi, CTRL, 0xF, 0xF, false);
  return fmaxf(x, __int_as_float(yi));
}
__device__ __forceinline__ float rowmax16(float x) {
  x = rowmax_step<0x121>(x);   // row_ror:1
  x = rowmax_step<0x122>(x);   // row_ror:2
  x = rowmax_step<0x124>(x);   // row_ror:4
  x = rowmax_step<0x128>(x);   // row_ror:8
  return x;
}

// ---------------------------------------------------------------------------
// Main kernel: one workgroup per (block b, head h, batch n).
// 8 waves; wave w computes rows [w*16, w*16+16) of the 128-row Q tile.
// Key window = 13 chunks of 32 keys:
//   chunk 0 : slot0 = global key (t=0, mask 0), slots 1..31 masked out
//   chunk c>=1 : keys t = (b-1)*128 + (c-1)*32 + j ; padding & t==0 masked
// Flash-style online softmax, f16 WMMA with fp32 accumulation.
// Row sums via an extra "ones column" WMMA tile (acc[4]).
// Double-buffered K/V chunks -> one barrier per chunk; reg prefetch overlaps
// global loads with compute.
// ---------------------------------------------------------------------------
__global__ __launch_bounds__(256) void attn_main(
    const float* __restrict__ Q, const float* __restrict__ K,
    const float* __restrict__ V, const float* __restrict__ AM,
    float* __restrict__ Out)
{
  __shared__ _Float16 Qs[BQ][72];        // Q tile, f16 (rows 144B, 16B-aligned)
  __shared__ _Float16 Ks[2][32][72];     // K chunk [buf][key][d]
  __shared__ _Float16 Vts[2][DH][40];    // V chunk transposed [buf][d][key]
  __shared__ float    Ms[2][32];         // per-key additive mask
  __shared__ _Float16 Ps[8][16][40];     // per-wave prob tile [row][key]

  const int b    = blockIdx.x;
  const int hh   = blockIdx.y;
  const int nn   = blockIdx.z;
  const int tid  = threadIdx.x;
  const int lane = tid & 31;
  const int wave = tid >> 5;
  const int half = lane >> 4;
  const int n16  = lane & 15;

  const size_t headoff = ((size_t)(nn * 16 + hh)) * TSEQ * DH;
  const float* Qg = Q + headoff + (unsigned)(b * BQ * DH);
  const float* Kg = K + headoff;
  const float* Vg = V + headoff;
  const float* Mg = AM + (unsigned)(nn * TSEQ);

  // loader role for this thread
  const int lr  = tid >> 3;        // key slot 0..31
  const int lc8 = (tid & 7) * 8;   // dim offset

  // ---- prefetch chunk 0 into registers (32-bit offsets -> saddr form) ----
  float4 pk0, pk1, pv0, pv1; float pmv = NEGF;
  {
    const unsigned off = (unsigned)(lr * DH + lc8);   // slot j -> key j (slot 0 = global key)
    pk0 = *reinterpret_cast<const float4*>(Kg + off);
    pk1 = *reinterpret_cast<const float4*>(Kg + off + 4);
    pv0 = *reinterpret_cast<const float4*>(Vg + off);
    pv1 = *reinterpret_cast<const float4*>(Vg + off + 4);
    if ((tid & 7) == 0) pmv = (lr == 0) ? 0.0f : NEGF;
  }

  // ---- preload Q tile f32 -> f16 in LDS ----
  for (int i = tid; i < BQ * DH / 8; i += 256) {
    int r  = i >> 3;
    int c8 = (i & 7) * 8;
    const float* src = Qg + (unsigned)(r * DH + c8);
    float4 a0 = *reinterpret_cast<const float4*>(src);
    float4 a1 = *reinterpret_cast<const float4*>(src + 4);
    float qa[8] = {a0.x, a0.y, a0.z, a0.w, a1.x, a1.y, a1.z, a1.w};
#pragma unroll
    for (int q = 0; q < 8; q++) Qs[r][c8 + q] = (_Float16)qa[q];
  }

  // ---- commit chunk 0 to buffer 0 ----
  {
    float kk[8] = {pk0.x, pk0.y, pk0.z, pk0.w, pk1.x, pk1.y, pk1.z, pk1.w};
    float vv[8] = {pv0.x, pv0.y, pv0.z, pv0.w, pv1.x, pv1.y, pv1.z, pv1.w};
#pragma unroll
    for (int q = 0; q < 8; q++) {
      Ks[0][lr][lc8 + q]  = (_Float16)kk[q];
      Vts[0][lc8 + q][lr] = (_Float16)vv[q];
    }
    if ((tid & 7) == 0) Ms[0][lr] = pmv;
  }
  __syncthreads();

  // ---- hoist chunk-invariant Q A-fragments (d 0..31 and 32..63) ----
  HV aQ[2];
#pragma unroll
  for (int dk = 0; dk < 2; dk++) {
    const _Float16* qrow = &Qs[wave * 16 + n16][dk * 32];
    aQ[dk].h[0] = *reinterpret_cast<const v8h*>(qrow + 8 * half);
    aQ[dk].h[1] = *reinterpret_cast<const v8h*>(qrow + 16 + 8 * half);
  }

  // ---- constant "ones column" B fragment: B[k][n] = (n==0) ----
  HV bones;
  {
    _Float16 e = (n16 == 0) ? (_Float16)1.0f : (_Float16)0.0f;
#pragma unroll
    for (int i = 0; i < 16; i++) bones.v[i] = e;
  }

  // ---- per-row online-softmax state (slot v <-> row v + 8*half) ----
  float mrow[8];
  v8f acc[5] = {};   // acc[0..3] = output col tiles, acc[4] = row-sum tile
#pragma unroll
  for (int v = 0; v < 8; v++) mrow[v] = NEGF;

  for (int c = 0; c < 13; c++) {
    const int cur = c & 1;

    // ---- issue next chunk's global loads (overlap with compute) ----
    if (c < 12) {
      int t0 = (b - 1) * BQ + c * 32 + lr;        // chunk c+1 keys
      bool valid = (t0 >= 0) && (t0 < TSEQ);
      unsigned off = (unsigned)((valid ? t0 : 0) * DH + lc8);
      pk0 = *reinterpret_cast<const float4*>(Kg + off);
      pk1 = *reinterpret_cast<const float4*>(Kg + off + 4);
      pv0 = *reinterpret_cast<const float4*>(Vg + off);
      pv1 = *reinterpret_cast<const float4*>(Vg + off + 4);
      if ((tid & 7) == 0)
        pmv = valid ? ((t0 == 0) ? NEGF : Mg[(unsigned)t0]) : NEGF;
    }

    // ---- S = Q * K^T : two 16x16 C tiles (keys 0-15 / 16-31) ----
    v8f S0 = {}, S1 = {};
#pragma unroll
    for (int dk = 0; dk < 2; dk++) {
      HV b0, b1;
      const _Float16* kr0 = &Ks[cur][n16][dk * 32 + 8 * half];
      const _Float16* kr1 = &Ks[cur][16 + n16][dk * 32 + 8 * half];
      b0.h[0] = *reinterpret_cast<const v8h*>(kr0);
      b0.h[1] = *reinterpret_cast<const v8h*>(kr0 + 16);
      b1.h[0] = *reinterpret_cast<const v8h*>(kr1);
      b1.h[1] = *reinterpret_cast<const v8h*>(kr1 + 16);
      S0 = __builtin_amdgcn_wmma_f32_16x16x32_f16(false, aQ[dk].v, false, b0.v, (short)0, S0, false, false);
      S1 = __builtin_amdgcn_wmma_f32_16x16x32_f16(false, aQ[dk].v, false, b1.v, (short)0, S1, false, false);
    }

    // ---- scale + mask; row max via DPP (VALU only) ----
    float mk0 = Ms[cur][n16], mk1 = Ms[cur][16 + n16];
    float s0[8], s1[8], corr[8];
#pragma unroll
    for (int v = 0; v < 8; v++) {
      s0[v] = S0[v] * SCALE + mk0;
      s1[v] = S1[v] * SCALE + mk1;
      float cm = rowmax16(fmaxf(s0[v], s1[v]));
      float mnew = fmaxf(mrow[v], cm);
      corr[v] = __expf(mrow[v] - mnew);
      mrow[v] = mnew;
      s0[v] = __expf(s0[v] - mnew);
      s1[v] = __expf(s1[v] - mnew);
    }
#pragma unroll
    for (int v = 0; v < 8; v++) {
#pragma unroll
      for (int j = 0; j < 5; j++) acc[j][v] *= corr[v];
      Ps[wave][v + 8 * half][n16]      = (_Float16)s0[v];   // C -> A re-layout via LDS
      Ps[wave][v + 8 * half][16 + n16] = (_Float16)s1[v];
    }

    // ---- O += P(16x32) * V(32x64), plus row-sum tile ----
    HV ap;
    const _Float16* prow = &Ps[wave][n16][0];
    ap.h[0] = *reinterpret_cast<const v8h*>(prow + 8 * half);
    ap.h[1] = *reinterpret_cast<const v8h*>(prow + 16 + 8 * half);
#pragma unroll
    for (int j = 0; j < 4; j++) {
      HV bv;
      const _Float16* vtr = &Vts[cur][j * 16 + n16][8 * half];
      bv.h[0] = *reinterpret_cast<const v8h*>(vtr);
      bv.h[1] = *reinterpret_cast<const v8h*>(vtr + 16);
      acc[j] = __builtin_amdgcn_wmma_f32_16x16x32_f16(false, ap.v, false, bv.v, (short)0, acc[j], false, false);
    }
    acc[4] = __builtin_amdgcn_wmma_f32_16x16x32_f16(false, ap.v, false, bones.v, (short)0, acc[4], false, false);

    // ---- commit prefetched chunk c+1 into the other buffer ----
    if (c < 12) {
      const int nxt = cur ^ 1;
      float kk[8] = {pk0.x, pk0.y, pk0.z, pk0.w, pk1.x, pk1.y, pk1.z, pk1.w};
      float vv[8] = {pv0.x, pv0.y, pv0.z, pv0.w, pv1.x, pv1.y, pv1.z, pv1.w};
#pragma unroll
      for (int q = 0; q < 8; q++) {
        Ks[nxt][lr][lc8 + q]  = (_Float16)kk[q];
        Vts[nxt][lc8 + q][lr] = (_Float16)vv[q];
      }
      if ((tid & 7) == 0) Ms[nxt][lr] = pmv;
    }
    __syncthreads();   // single barrier per chunk (double-buffered)
  }

  // ---- epilogue: broadcast row sums (col 0 of ones tile), normalize, store ----
  float* ob = Out + headoff + (unsigned)(b * BQ * DH);
#pragma unroll
  for (int v = 0; v < 8; v++) {
    float lsum = __shfl(acc[4][v], 16 * half, 32);   // column-0 lane of this row group
    float inv = 1.0f / lsum;
    int row = wave * 16 + v + 8 * half;
#pragma unroll
    for (int j = 0; j < 4; j++)
      ob[(unsigned)(row * DH + j * 16 + n16)] = acc[j][v] * inv;
  }
}

// ---------------------------------------------------------------------------
// Fixup kernel: query t=0 attends over ALL 4096 keys (mask[0] = -inf).
// One workgroup per (n,h). Overwrites out row 0 of that head.
// ---------------------------------------------------------------------------
__global__ __launch_bounds__(256) void attn_row0(
    const float* __restrict__ Q, const float* __restrict__ K,
    const float* __restrict__ V, const float* __restrict__ AM,
    float* __restrict__ Out)
{
  __shared__ float q0[DH];
  __shared__ float sc[TSEQ];
  __shared__ float red[256];
  __shared__ float part[4][DH];

  const int head = blockIdx.x;           // 0..31 = n*16+h
  const int nn   = head / 16;
  const size_t ho = (size_t)head * TSEQ * DH;
  const int tid = threadIdx.x;

  if (tid < DH) q0[tid] = Q[ho + tid];
  __syncthreads();

  float lmax = NEGF;
  for (int k = tid; k < TSEQ; k += 256) {
    const float* kr = K + ho + (unsigned)(k * DH);
    float s = 0.0f;
#pragma unroll 16
    for (int d = 0; d < DH; d++) s += q0[d] * kr[d];
    s = s * SCALE + ((k == 0) ? NEGF : AM[(unsigned)(nn * TSEQ + k)]);
    sc[k] = s;
    lmax = fmaxf(lmax, s);
  }
  red[tid] = lmax;
  __syncthreads();
  for (int o = 128; o > 0; o >>= 1) {
    if (tid < o) red[tid] = fmaxf(red[tid], red[tid + o]);
    __syncthreads();
  }
  const float M = red[0];
  __syncthreads();

  float lsum = 0.0f;
  for (int k = tid; k < TSEQ; k += 256) {
    float p = __expf(sc[k] - M);
    sc[k] = p;
    lsum += p;
  }
  red[tid] = lsum;
  __syncthreads();
  for (int o = 128; o > 0; o >>= 1) {
    if (tid < o) red[tid] += red[tid + o];
    __syncthreads();
  }
  const float L = red[0];
  __syncthreads();

  const int d = tid & 63;
  const int g = tid >> 6;                 // 4 key groups of 1024
  float a = 0.0f;
  for (int k = g * 1024; k < (g + 1) * 1024; k++)
    a += sc[k] * V[ho + (unsigned)(k * DH + d)];
  part[g][d] = a;
  __syncthreads();
  if (tid < DH)
    Out[ho + tid] = (part[0][tid] + part[1][tid] + part[2][tid] + part[3][tid]) / L;
}

// ---------------------------------------------------------------------------
extern "C" void kernel_launch(void* const* d_in, const int* in_sizes, int n_in,
                              void* d_out, int out_size, void* d_ws, size_t ws_size,
                              hipStream_t stream) {
  (void)in_sizes; (void)n_in; (void)out_size; (void)d_ws; (void)ws_size;
  const float* Q  = (const float*)d_in[0];
  const float* K  = (const float*)d_in[1];
  const float* V  = (const float*)d_in[2];
  const float* AM = (const float*)d_in[3];
  float* Out = (float*)d_out;

  dim3 grid(TSEQ / BQ, 16, 2);   // (blocks, heads, batch)
  attn_main<<<grid, 256, 0, stream>>>(Q, K, V, AM, Out);
  attn_row0<<<32, 256, 0, stream>>>(Q, K, V, AM, Out);
}
